// ParticleAttentionBlock_77833397338629
// MI455X (gfx1250) — matrix operand order repaired and checked by reading, not compiled, not run on hardware
//
#include <hip/hip_runtime.h>
#include <hip/hip_bf16.h>

// ---------------------------------------------------------------------------
// ParticleAttentionBlock for MI455X (gfx1250, wave32, WMMA)
// All GEMMs via v_wmma_f32_16x16x32_f16 (f32 accumulate).
// A-tile staging via gfx1250 global_load_async_to_lds_b128 (ASYNCcnt path).
// ---------------------------------------------------------------------------

typedef __attribute__((ext_vector_type(16))) _Float16 v16h;
typedef __attribute__((ext_vector_type(8)))  _Float16 v8h;
typedef __attribute__((ext_vector_type(8)))  float    v8f;
typedef __attribute__((vector_size(16)))     int      v4i;

#define BB 64
#define PP 256
#define EE 256
#define HH 8
#define HD 32
#define UU 6
#define DFF 1024
#define MTOT (BB * PP)      // 16384
#define NEGV -1.0e9f

#if defined(__has_builtin)
#if __has_builtin(__builtin_amdgcn_global_load_async_to_lds_b128) && \
    __has_builtin(__builtin_amdgcn_s_wait_asynccnt)
#define HAVE_ASYNC_LDS 1
#endif
#endif

__device__ inline v16h combine16(v8h lo, v8h hi) {
  v16h r;
#pragma unroll
  for (int i = 0; i < 8; ++i) { r[i] = lo[i]; r[i + 8] = hi[i]; }
  return r;
}

__device__ inline v8f wmma_f16f32(v16h a, v16h b, v8f c) {
  // D = A(16x32 f16) * B(32x16 f16) + C(16x16 f32)
  return __builtin_amdgcn_wmma_f32_16x16x32_f16(
      /*neg_a=*/false, a, /*neg_b=*/false, b,
      /*c_mod=*/(short)0, c, /*reuse_a=*/false, /*reuse_b=*/false);
}

// 16-byte global -> LDS copy; gfx1250 async path when the builtin exists.
__device__ inline void copy16_to_lds(const _Float16* g, _Float16* l) {
#ifdef HAVE_ASYNC_LDS
  __builtin_amdgcn_global_load_async_to_lds_b128(
      (__attribute__((address_space(1))) v4i*)(void*)g,
      (__attribute__((address_space(3))) v4i*)(void*)l, 0, 0);
#else
  *(v8h*)l = *(const v8h*)g;
#endif
}
__device__ inline void lds_copy_wait() {
#ifdef HAVE_ASYNC_LDS
  __builtin_amdgcn_s_wait_asynccnt(0);
#endif
}

// ---------------------------------------------------------------------------
// Weight transpose + convert: W[K,N] f32 -> WT[N,K] f16 (LDS tiled)
// ---------------------------------------------------------------------------
__global__ __launch_bounds__(256) void transpose_w_kernel(
    const float* __restrict__ W, _Float16* __restrict__ WT, int K, int N) {
  __shared__ float tile[32][33];
  int n0 = blockIdx.x * 32, k0 = blockIdx.y * 32;
  int tx = threadIdx.x & 31, ty = threadIdx.x >> 5;  // 32 x 8
#pragma unroll
  for (int i = ty; i < 32; i += 8)
    tile[i][tx] = W[(size_t)(k0 + i) * N + n0 + tx];
  __syncthreads();
#pragma unroll
  for (int i = ty; i < 32; i += 8)
    WT[(size_t)(n0 + i) * K + k0 + tx] = (_Float16)tile[tx][i];
}

// ---------------------------------------------------------------------------
// LayerNorm1: one wave per row of 256, 8 elems/lane, wave32 shuffles
// ---------------------------------------------------------------------------
__global__ __launch_bounds__(256) void layernorm1_kernel(
    const float* __restrict__ x, const float* __restrict__ w,
    const float* __restrict__ b, _Float16* __restrict__ xnh) {
  int lane = threadIdx.x & 31, wid = threadIdx.x >> 5;
  size_t row = (size_t)blockIdx.x * 8 + wid;
  const float* xr = x + row * EE;
  float vals[8], s = 0.f;
#pragma unroll
  for (int i = 0; i < 8; ++i) { vals[i] = xr[lane * 8 + i]; s += vals[i]; }
#pragma unroll
  for (int m = 16; m >= 1; m >>= 1) s += __shfl_xor(s, m, 32);
  float mean = s * (1.f / EE);
  float vs = 0.f;
#pragma unroll
  for (int i = 0; i < 8; ++i) { float d = vals[i] - mean; vs += d * d; }
#pragma unroll
  for (int m = 16; m >= 1; m >>= 1) vs += __shfl_xor(vs, m, 32);
  float inv = rsqrtf(vs * (1.f / EE) + 1e-5f);
#pragma unroll
  for (int i = 0; i < 8; ++i) {
    int c = lane * 8 + i;
    xnh[row * EE + c] = (_Float16)((vals[i] - mean) * inv * w[c] + b[c]);
  }
}

// ---------------------------------------------------------------------------
// LN2 + residual + RMSNorm: h = LN(ao)*w+b + x ; hn = h*rsqrt(mean(h^2)+eps)*rw
// ---------------------------------------------------------------------------
__global__ __launch_bounds__(256) void ln2_rms_kernel(
    const float* __restrict__ ao, const float* __restrict__ x,
    const float* __restrict__ w, const float* __restrict__ b,
    const float* __restrict__ rw, float* __restrict__ hOut,
    _Float16* __restrict__ hnh) {
  int lane = threadIdx.x & 31, wid = threadIdx.x >> 5;
  size_t row = (size_t)blockIdx.x * 8 + wid;
  const float* ar = ao + row * EE;
  float vals[8], s = 0.f;
#pragma unroll
  for (int i = 0; i < 8; ++i) { vals[i] = ar[lane * 8 + i]; s += vals[i]; }
#pragma unroll
  for (int m = 16; m >= 1; m >>= 1) s += __shfl_xor(s, m, 32);
  float mean = s * (1.f / EE);
  float vs = 0.f;
#pragma unroll
  for (int i = 0; i < 8; ++i) { float d = vals[i] - mean; vs += d * d; }
#pragma unroll
  for (int m = 16; m >= 1; m >>= 1) vs += __shfl_xor(vs, m, 32);
  float inv = rsqrtf(vs * (1.f / EE) + 1e-5f);
  float hv[8], s2 = 0.f;
#pragma unroll
  for (int i = 0; i < 8; ++i) {
    int c = lane * 8 + i;
    hv[i] = (vals[i] - mean) * inv * w[c] + b[c] + x[row * EE + c];
    hOut[row * EE + c] = hv[i];
    s2 += hv[i] * hv[i];
  }
#pragma unroll
  for (int m = 16; m >= 1; m >>= 1) s2 += __shfl_xor(s2, m, 32);
  float inv2 = rsqrtf(s2 * (1.f / EE) + 1.1920929e-07f);
#pragma unroll
  for (int i = 0; i < 8; ++i) {
    int c = lane * 8 + i;
    hnh[row * EE + c] = (_Float16)(hv[i] * inv2 * rw[c]);
  }
}

// ---------------------------------------------------------------------------
// WMMA GEMM: C[M,N] = A[M,K](f16) @ W[K,N](f16), W given transposed WT[N,K].
// Block tile 128x64, 8 waves arranged 4(M) x 2(N); each wave: 32x32 subtile,
// 4 WMMA per K-step. A tile staged in LDS (async-to-LDS on gfx1250); B
// fragments loaded directly from WT (contiguous, L2-resident weights).
// flags bit0 = SiLU. Optional bias, residual(f32), f32 and/or f16 outputs.
// ---------------------------------------------------------------------------
__global__ __launch_bounds__(256) void gemm_wmma_kernel(
    const _Float16* __restrict__ A, const _Float16* __restrict__ WT,
    const float* __restrict__ bias, const float* __restrict__ res,
    float* __restrict__ outF, _Float16* __restrict__ outH,
    int M, int N, int K, int flags) {
  (void)M;
  int tid = threadIdx.x, lane = tid & 31, wid = tid >> 5;
  int waveM = wid & 3, waveN = wid >> 2;
  int bm = blockIdx.y * 128, bn = blockIdx.x * 64;

  __shared__ _Float16 sA[128 * 32];  // A tile, row-major [128][32]

  v8f c00 = {}, c01 = {}, c10 = {}, c11 = {};
  int g = (lane >> 4) * 8;

  // A staging: each thread moves 32 bytes (2 x b128)
  int srow = tid >> 1, schunk = (tid & 1) * 16;
  const _Float16* aSrc = A + (size_t)(bm + srow) * K + schunk;
  _Float16* sDst = &sA[srow * 32 + schunk];

  for (int k0 = 0; k0 < K; k0 += 32) {
    copy16_to_lds(aSrc + k0, sDst);
    copy16_to_lds(aSrc + k0 + 8, sDst + 8);
    lds_copy_wait();
    __syncthreads();
    if (k0 + 32 < K)  // gfx1250 global_prefetch_b8 on next A tile
      __builtin_prefetch(aSrc + k0 + 32, 0, 1);

    const _Float16* ar0 = &sA[(waveM * 32 + (lane & 15)) * 32];
    const _Float16* ar1 = ar0 + 16 * 32;
    v16h a0 = combine16(*(const v8h*)(ar0 + 2 * g), *(const v8h*)(ar0 + 2 * g + 8));
    v16h a1 = combine16(*(const v8h*)(ar1 + 2 * g), *(const v8h*)(ar1 + 2 * g + 8));

    const _Float16* w0 =
        WT + (size_t)(bn + waveN * 32 + (lane & 15)) * K + k0 + (lane >> 4) * 16;
    const _Float16* w1 = w0 + (size_t)16 * K;
    v16h b0 = combine16(*(const v8h*)w0, *(const v8h*)(w0 + 8));
    v16h b1 = combine16(*(const v8h*)w1, *(const v8h*)(w1 + 8));

    c00 = wmma_f16f32(a0, b0, c00);
    c01 = wmma_f16f32(a0, b1, c01);
    c10 = wmma_f16f32(a1, b0, c10);
    c11 = wmma_f16f32(a1, b1, c11);
    __syncthreads();
  }

  // epilogue: C layout — VGPR v -> row m = v + 8*(lane/16); col n = lane%16
#pragma unroll
  for (int ti = 0; ti < 2; ++ti) {
#pragma unroll
    for (int tj = 0; tj < 2; ++tj) {
      v8f acc = (ti == 0) ? ((tj == 0) ? c00 : c01) : ((tj == 0) ? c10 : c11);
#pragma unroll
      for (int v = 0; v < 8; ++v) {
        int gr = bm + waveM * 32 + ti * 16 + v + 8 * (lane >> 4);
        int gc = bn + waveN * 32 + tj * 16 + (lane & 15);
        float val = acc[v];
        if (bias) val += bias[gc];
        if (flags & 1) val = val / (1.f + __expf(-val));  // SiLU
        size_t idx = (size_t)gr * N + gc;
        if (res) val += res[idx];
        if (outF) outF[idx] = val;
        if (outH) outH[idx] = (_Float16)val;
      }
    }
  }
}

// ---------------------------------------------------------------------------
// uh = einsum('bupq,hu->bhpq') + ub ; top-32 mask over sum_h |uh| ; NEG fill.
// One block per (b,p) row; exact top-k with lower-index tie-break (jax top_k).
// ---------------------------------------------------------------------------
__global__ __launch_bounds__(256) void uh_topk_kernel(
    const float* __restrict__ u, const float* __restrict__ uw,
    const float* __restrict__ ub, float* __restrict__ uhm) {
  int p = blockIdx.x, b = blockIdx.y;
  int t = threadIdx.x;
  float uv[UU];
#pragma unroll
  for (int q = 0; q < UU; ++q)
    uv[q] = u[(((size_t)b * UU + q) * PP + p) * PP + t];
  float uhv[HH];
  float imp = 0.f;
#pragma unroll
  for (int h = 0; h < HH; ++h) {
    float a = ub[h];
#pragma unroll
    for (int q = 0; q < UU; ++q) a += uw[h * UU + q] * uv[q];
    uhv[h] = a;
    imp += fabsf(a);
  }

  __shared__ float sv[256];
  __shared__ int si[256];
  float my = imp;
  bool sel = false;
  for (int iter = 0; iter < 32; ++iter) {
    sv[t] = my; si[t] = t;
    __syncthreads();
    for (int s = 128; s > 0; s >>= 1) {
      if (t < s) {
        if (sv[t + s] > sv[t] || (sv[t + s] == sv[t] && si[t + s] < si[t])) {
          sv[t] = sv[t + s]; si[t] = si[t + s];
        }
      }
      __syncthreads();
    }
    int winner = si[0];
    __syncthreads();
    if (t == winner) { sel = true; my = -INFINITY; }
  }
#pragma unroll
  for (int h = 0; h < HH; ++h)
    uhm[(((size_t)b * HH + h) * PP + p) * PP + t] = sel ? uhv[h] : NEGV;
}

// ---------------------------------------------------------------------------
// Attention: one block per (b,h). 8 waves * 2 passes cover 256 q rows.
// scores via WMMA (K=HD=32), +uh, softmax (half-wave shuffles), attn -> d_out,
// ctx = attn @ v via WMMA with LDS-transposed V and LDS-staged attn fragments.
// ---------------------------------------------------------------------------
__global__ __launch_bounds__(256) void attention_kernel(
    const _Float16* __restrict__ qh, const _Float16* __restrict__ kh,
    const _Float16* __restrict__ vh, const float* __restrict__ uhm,
    float* __restrict__ attn_out, _Float16* __restrict__ ctxh) {
  int bh = blockIdx.x;
  int b = bh >> 3, h = bh & 7;
  int tid = threadIdx.x, lane = tid & 31, wid = tid >> 5;
  const int VS = 264;  // padded stride for vT rows (16B-aligned)

  __shared__ _Float16 vT[HD * VS];        // V transposed: [d][key]
  __shared__ _Float16 stage[8 * 16 * 32]; // per-wave attn f16 staging

  // load V head [256, 32] transposed into LDS
  {
    int key = tid;
    const _Float16* vr = vh + ((size_t)(b * PP + key) * EE + h * HD);
#pragma unroll
    for (int d = 0; d < HD; ++d) vT[d * VS + key] = vr[d];
  }
  __syncthreads();

  const float scale = 0.1767766952966369f;  // HD^-0.5
  int g = (lane >> 4) * 8;

  for (int pass = 0; pass < 2; ++pass) {
    int qbase = (pass * 8 + wid) * 16;
    // A fragment: q tile [16 rows, 32]
    const _Float16* qrow =
        qh + ((size_t)(b * PP + qbase + (lane & 15)) * EE + h * HD);
    v16h aq = combine16(*(const v8h*)(qrow + 2 * g), *(const v8h*)(qrow + 2 * g + 8));

    v8f s[16];
#pragma unroll
    for (int t = 0; t < 16; ++t) {
      // B fragment: k^T tile; lane col = t*16 + lane%16, 16 contiguous halves
      const _Float16* krow =
          kh + ((size_t)(b * PP + t * 16 + (lane & 15)) * EE + h * HD) +
          (lane >> 4) * 16;
      v16h bk = combine16(*(const v8h*)krow, *(const v8h*)(krow + 8));
      v8f z = {};
      s[t] = wmma_f16f32(aq, bk, z);
    }

    // scale + uh + row softmax (rows live within 16-lane halves)
#pragma unroll
    for (int v = 0; v < 8; ++v) {
      int row = qbase + v + 8 * (lane >> 4);
      const float* uhr = &uhm[(((size_t)b * HH + h) * PP + row) * PP];
      float mx = -INFINITY;
#pragma unroll
      for (int t = 0; t < 16; ++t) {
        float val = s[t][v] * scale + uhr[t * 16 + (lane & 15)];
        s[t][v] = val;
        mx = fmaxf(mx, val);
      }
#pragma unroll
      for (int m = 8; m >= 1; m >>= 1) mx = fmaxf(mx, __shfl_xor(mx, m, 16));
      float sum = 0.f;
#pragma unroll
      for (int t = 0; t < 16; ++t) {
        float e = __expf(s[t][v] - mx);
        s[t][v] = e;
        sum += e;
      }
#pragma unroll
      for (int m = 8; m >= 1; m >>= 1) sum += __shfl_xor(sum, m, 16);
      float inv = (sum > 0.f) ? (1.f / sum) : 0.f;  // nan_to_num
#pragma unroll
      for (int t = 0; t < 16; ++t) s[t][v] *= inv;
      // write attn output (f32, required by reference)
      float* ar = &attn_out[(((size_t)b * HH + h) * PP + row) * PP];
#pragma unroll
      for (int t = 0; t < 16; ++t) ar[t * 16 + (lane & 15)] = s[t][v];
    }

    // ctx = attn @ v : loop over key blocks of 32 (K of the WMMA)
    v8f cc0 = {}, cc1 = {};
    for (int kk = 0; kk < 8; ++kk) {
#pragma unroll
      for (int v = 0; v < 8; ++v) {
        int m = v + 8 * (lane >> 4);
        stage[wid * 512 + m * 32 + (lane & 15)] = (_Float16)s[2 * kk][v];
        stage[wid * 512 + m * 32 + 16 + (lane & 15)] = (_Float16)s[2 * kk + 1][v];
      }
      __syncthreads();
      const _Float16* ar = &stage[wid * 512 + (lane & 15) * 32];
      v16h af = combine16(*(const v8h*)(ar + 2 * g), *(const v8h*)(ar + 2 * g + 8));
      const _Float16* b0p = &vT[(lane & 15) * VS + kk * 32 + (lane >> 4) * 16];
      v16h bv0 = combine16(*(const v8h*)b0p, *(const v8h*)(b0p + 8));
      const _Float16* b1p =
          &vT[(16 + (lane & 15)) * VS + kk * 32 + (lane >> 4) * 16];
      v16h bv1 = combine16(*(const v8h*)b1p, *(const v8h*)(b1p + 8));
      cc0 = wmma_f16f32(af, bv0, cc0);
      cc1 = wmma_f16f32(af, bv1, cc1);
      __syncthreads();
    }
#pragma unroll
    for (int v = 0; v < 8; ++v) {
      int row = qbase + v + 8 * (lane >> 4);
      _Float16* cr = ctxh + ((size_t)(b * PP + row) * EE + h * HD);
      cr[lane & 15] = (_Float16)cc0[v];
      cr[16 + (lane & 15)] = (_Float16)cc1[v];
    }
  }
}

// ---------------------------------------------------------------------------
// Host-side orchestration
// ---------------------------------------------------------------------------
extern "C" void kernel_launch(void* const* d_in, const int* in_sizes, int n_in,
                              void* d_out, int out_size, void* d_ws,
                              size_t ws_size, hipStream_t stream) {
  (void)in_sizes; (void)n_in; (void)out_size; (void)ws_size;
  const float* x      = (const float*)d_in[0];
  const float* u      = (const float*)d_in[1];
  const float* n1w    = (const float*)d_in[2];
  const float* n1b    = (const float*)d_in[3];
  const float* wq     = (const float*)d_in[4];
  const float* wk     = (const float*)d_in[5];
  const float* wv     = (const float*)d_in[6];
  const float* wo     = (const float*)d_in[7];
  const float* bo     = (const float*)d_in[8];
  const float* uw     = (const float*)d_in[9];
  const float* ub     = (const float*)d_in[10];
  const float* n2w    = (const float*)d_in[11];
  const float* n2b    = (const float*)d_in[12];
  const float* rmsw   = (const float*)d_in[13];
  const float* w1     = (const float*)d_in[14];
  const float* b1     = (const float*)d_in[15];
  const float* w2     = (const float*)d_in[16];
  const float* b2     = (const float*)d_in[17];

  char* ws = (char*)d_ws;
  size_t off = 0;
  auto alloc = [&](size_t bytes) -> void* {
    void* p = ws + off;
    off = (off + bytes + 255) & ~(size_t)255;
    return p;
  };

  _Float16* xnh = (_Float16*)alloc((size_t)MTOT * EE * 2);
  _Float16* wqT = (_Float16*)alloc((size_t)EE * EE * 2);   // [N,K] f16
  _Float16* wkT = (_Float16*)alloc((size_t)EE * EE * 2);
  _Float16* wvT = (_Float16*)alloc((size_t)EE * EE * 2);
  _Float16* woT = (_Float16*)alloc((size_t)EE * EE * 2);
  _Float16* w1T = (_Float16*)alloc((size_t)EE * DFF * 2);
  _Float16* w2T = (_Float16*)alloc((size_t)DFF * EE * 2);
  _Float16* qf  = (_Float16*)alloc((size_t)MTOT * EE * 2);
  _Float16* kf  = (_Float16*)alloc((size_t)MTOT * EE * 2);
  _Float16* vf  = (_Float16*)alloc((size_t)MTOT * EE * 2);
  float*    uhm = (float*)alloc((size_t)BB * HH * PP * PP * 4);
  _Float16* ctxh= (_Float16*)alloc((size_t)MTOT * EE * 2);
  float*    ao  = (float*)alloc((size_t)MTOT * EE * 4);
  float*    hbuf= (float*)alloc((size_t)MTOT * EE * 4);
  _Float16* hnh = (_Float16*)alloc((size_t)MTOT * EE * 2);
  _Float16* t1h = (_Float16*)alloc((size_t)MTOT * DFF * 2);

  float* zOut = (float*)d_out;                       // [B,P,E]
  float* attnOut = zOut + (size_t)BB * PP * EE;      // [B,H,P,P]

  // weight transpose + f16 convert: W[K,N] -> WT[N,K]
  transpose_w_kernel<<<dim3(EE / 32, EE / 32), 256, 0, stream>>>(wq, wqT, EE, EE);
  transpose_w_kernel<<<dim3(EE / 32, EE / 32), 256, 0, stream>>>(wk, wkT, EE, EE);
  transpose_w_kernel<<<dim3(EE / 32, EE / 32), 256, 0, stream>>>(wv, wvT, EE, EE);
  transpose_w_kernel<<<dim3(EE / 32, EE / 32), 256, 0, stream>>>(wo, woT, EE, EE);
  transpose_w_kernel<<<dim3(DFF / 32, EE / 32), 256, 0, stream>>>(w1, w1T, EE, DFF);
  transpose_w_kernel<<<dim3(EE / 32, DFF / 32), 256, 0, stream>>>(w2, w2T, DFF, EE);

  // LayerNorm 1
  layernorm1_kernel<<<MTOT / 8, 256, 0, stream>>>(x, n1w, n1b, xnh);

  // Q/K/V projections (WMMA)
  dim3 gEE(EE / 64, MTOT / 128);
  gemm_wmma_kernel<<<gEE, 256, 0, stream>>>(xnh, wqT, nullptr, nullptr,
                                            nullptr, qf, MTOT, EE, EE, 0);
  gemm_wmma_kernel<<<gEE, 256, 0, stream>>>(xnh, wkT, nullptr, nullptr,
                                            nullptr, kf, MTOT, EE, EE, 0);
  gemm_wmma_kernel<<<gEE, 256, 0, stream>>>(xnh, wvT, nullptr, nullptr,
                                            nullptr, vf, MTOT, EE, EE, 0);

  // pairwise bias + top-k mask
  uh_topk_kernel<<<dim3(PP, BB), 256, 0, stream>>>(u, uw, ub, uhm);

  // attention (scores, softmax, attn -> d_out, ctx)
  attention_kernel<<<BB * HH, 256, 0, stream>>>(qf, kf, vf, uhm, attnOut, ctxh);

  // output projection + bo
  gemm_wmma_kernel<<<gEE, 256, 0, stream>>>(ctxh, woT, bo, nullptr, ao,
                                            nullptr, MTOT, EE, EE, 0);

  // LN2 + residual + RMSNorm
  ln2_rms_kernel<<<MTOT / 8, 256, 0, stream>>>(ao, x, n2w, n2b, rmsw, hbuf, hnh);

  // MLP: SiLU(hn@w1 + b1) @ w2 + b2 + h  -> z + h
  dim3 gFF(DFF / 64, MTOT / 128);
  gemm_wmma_kernel<<<gFF, 256, 0, stream>>>(hnh, w1T, b1, nullptr, nullptr,
                                            t1h, MTOT, DFF, EE, 1);
  gemm_wmma_kernel<<<gEE, 256, 0, stream>>>(t1h, w2T, b2, hbuf, zOut,
                                            nullptr, MTOT, EE, DFF, 0);
}
